// GAT_14053132992853
// MI455X (gfx1250) — compile-verified, matrix operand order checked by reading, not run on hardware
//
#include <hip/hip_runtime.h>

// ---------------------------------------------------------------------------
// GAT (2-layer, 8 heads) for MI455X / gfx1250.
// GEMMs: v_wmma_f32_16x16x32_bf16, 32x64 register-blocked macro-tile per wave,
// double-buffered fragment loads. Graph part: float atomics, float4 gather.
// ---------------------------------------------------------------------------

#define LANES 32
#define HEADS 8
#define SLOPE 0.2f
#define MT 2   // 16-row tiles per wave
#define NT 4   // 16-col tiles per wave

typedef __attribute__((ext_vector_type(16))) __bf16 bf16x16;
typedef __attribute__((ext_vector_type(8)))  float  floatx8;

union BfFrag { unsigned int u[8]; bf16x16 v; };

// ---- helpers ---------------------------------------------------------------

__device__ __forceinline__ unsigned short f2bf(float f) {
    unsigned int u = __float_as_uint(f);
    u += 0x7FFFu + ((u >> 16) & 1u);   // round-to-nearest-even
    return (unsigned short)(u >> 16);
}

__device__ __forceinline__ void atomicMaxFloat(float* addr, float val) {
    if (val >= 0.0f)
        atomicMax((int*)addr, __float_as_int(val));
    else
        atomicMin((unsigned int*)addr, __float_as_uint(val));
}

__device__ __forceinline__ float lrelu(float v) {
    return v > 0.0f ? v : SLOPE * v;
}

__device__ __forceinline__ void load_frag8(BfFrag& dst, const unsigned int* row,
                                           int kq, const int* kd) {
#pragma unroll
    for (int v = 0; v < 8; ++v) dst.u[v] = row[kq + kd[v]];
}

// ---- elementwise kernels ---------------------------------------------------

__global__ void gat_fill_f32(float* __restrict__ p, float v, long long n) {
    long long i = (long long)blockIdx.x * blockDim.x + threadIdx.x;
    if (i < n) p[i] = v;
}

__global__ void gat_cast_bf16(const float* __restrict__ in,
                              unsigned short* __restrict__ out, long long n) {
    long long i = (long long)blockIdx.x * blockDim.x + threadIdx.x;
    if (i < n) out[i] = f2bf(in[i]);
}

// W[k][nc] (row-major, K x Nc) -> Wt[nc][k] bf16
__global__ void gat_transpose_cast(const float* __restrict__ W,
                                   unsigned short* __restrict__ Wt,
                                   int K, int Nc) {
    long long i = (long long)blockIdx.x * blockDim.x + threadIdx.x;
    if (i >= (long long)K * Nc) return;
    int k = (int)(i / Nc);
    int n = (int)(i % Nc);
    Wt[(size_t)n * K + k] = f2bf(W[(size_t)k * Nc + n]);
}

// relu(acc + bias) -> bf16 (layer-1 output -> layer-2 input)
__global__ void gat_relu_bias_cast(const float* __restrict__ acc,
                                   const float* __restrict__ bias, int C,
                                   unsigned short* __restrict__ out,
                                   long long n) {
    long long i = (long long)blockIdx.x * blockDim.x + threadIdx.x;
    if (i >= n) return;
    float v = acc[i] + bias[i % C];
    out[i] = f2bf(v > 0.0f ? v : 0.0f);
}

// ---- WMMA GEMM: C[M x Nc] = A[M x K](bf16) @ Bt[Nc x K](bf16, col-major) ---
// Each wave computes a 32x64 macro-tile (MT x NT 16x16 tiles), K-loop
// double-buffered so fragment loads for step k+32 overlap WMMAs of step k.

__global__ void __launch_bounds__(256)
gat_wmma_gemm(const unsigned short* __restrict__ A,
              const unsigned short* __restrict__ Bt,
              float* __restrict__ C, int M, int K, int Nc) {
    const int lane = threadIdx.x & (LANES - 1);
    const int wave = threadIdx.x >> 5;
    const int nMacs = Nc / (16 * NT);
    const long long wl = (long long)blockIdx.x * 8 + wave;
    const long long total = (long long)(M / (16 * MT)) * nMacs;
    if (wl >= total) return;
    const int mMac = (int)(wl / nMacs);
    const int nMac = (int)(wl % nMacs);

    const int hi = lane >> 4;       // K-half selector (lanes 0-15 / 16-31)
    const int lr = lane & 15;

    const unsigned int* arow[MT];
    const unsigned int* brow[NT];
#pragma unroll
    for (int i = 0; i < MT; ++i)
        arow[i] = (const unsigned int*)(A + (size_t)(mMac * 16 * MT + i * 16 + lr) * K);
#pragma unroll
    for (int j = 0; j < NT; ++j)
        brow[j] = (const unsigned int*)(Bt + (size_t)(nMac * 16 * NT + j * 16 + lr) * K);

    // ISA 16-bit A/B packing, as dword (K-pair) indices within a 32-K step:
    // {0,1,2,3, 8,9,10,11} + hi*4 -> coalesces to 2x global_load_b128.
    int kd[8];
#pragma unroll
    for (int v = 0; v < 8; ++v)
        kd[v] = (hi << 2) + ((v < 4) ? v : (4 + v));

    BfFrag a0[MT], b0[NT], a1[MT], b1[NT];
    floatx8 acc[MT][NT] = {};

#pragma unroll
    for (int i = 0; i < MT; ++i) load_frag8(a0[i], arow[i], 0, kd);
#pragma unroll
    for (int j = 0; j < NT; ++j) load_frag8(b0[j], brow[j], 0, kd);

    for (int k0 = 0; k0 < K; k0 += 64) {
        // load buffer 1 (k0+32) while computing buffer 0 (k0)
        if (k0 + 32 < K) {
#pragma unroll
            for (int i = 0; i < MT; ++i) load_frag8(a1[i], arow[i], (k0 + 32) >> 1, kd);
#pragma unroll
            for (int j = 0; j < NT; ++j) load_frag8(b1[j], brow[j], (k0 + 32) >> 1, kd);
        }
#pragma unroll
        for (int i = 0; i < MT; ++i)
#pragma unroll
            for (int j = 0; j < NT; ++j)
                acc[i][j] = __builtin_amdgcn_wmma_f32_16x16x32_bf16(
                    false, a0[i].v, false, b0[j].v, (short)0, acc[i][j], false, false);

        // load buffer 0 (k0+64) while computing buffer 1 (k0+32)
        if (k0 + 64 < K) {
#pragma unroll
            for (int i = 0; i < MT; ++i) load_frag8(a0[i], arow[i], (k0 + 64) >> 1, kd);
#pragma unroll
            for (int j = 0; j < NT; ++j) load_frag8(b0[j], brow[j], (k0 + 64) >> 1, kd);
        }
        if (k0 + 32 < K) {
#pragma unroll
            for (int i = 0; i < MT; ++i)
#pragma unroll
                for (int j = 0; j < NT; ++j)
                    acc[i][j] = __builtin_amdgcn_wmma_f32_16x16x32_bf16(
                        false, a1[i].v, false, b1[j].v, (short)0, acc[i][j], false, false);
        }
    }

    // D layout: VGPR r holds row base + hi*8 + r, col = base + lr
#pragma unroll
    for (int i = 0; i < MT; ++i)
#pragma unroll
        for (int j = 0; j < NT; ++j)
#pragma unroll
            for (int r = 0; r < 8; ++r) {
                int row = mMac * 16 * MT + i * 16 + hi * 8 + r;
                int col = nMac * 16 * NT + j * 16 + lr;
                C[(size_t)row * Nc + col] = acc[i][j][r];
            }
}

// ---- attention scores: el/er[n][h] = sum_d feat[n][h][d] * a{l,r}[h][d] ----

__global__ void __launch_bounds__(256)
gat_node_scores(const float* __restrict__ feat,
                const float* __restrict__ al, const float* __restrict__ ar,
                float* __restrict__ el, float* __restrict__ er,
                int N, int D) {
    const int lane = threadIdx.x & (LANES - 1);
    const int node = blockIdx.x * 8 + (threadIdx.x >> 5);
    if (node >= N) return;
    const int C = HEADS * D;
    for (int h = 0; h < HEADS; ++h) {
        float sl = 0.0f, sr = 0.0f;
        for (int d = lane; d < D; d += LANES) {
            float f = feat[(size_t)node * C + h * D + d];
            sl += f * al[h * D + d];
            sr += f * ar[h * D + d];
        }
#pragma unroll
        for (int msk = 16; msk >= 1; msk >>= 1) {
            sl += __shfl_xor(sl, msk, LANES);
            sr += __shfl_xor(sr, msk, LANES);
        }
        if (lane == 0) {
            el[(size_t)node * HEADS + h] = sl;
            er[(size_t)node * HEADS + h] = sr;
        }
    }
}

// ---- edge passes -----------------------------------------------------------

__global__ void gat_edge_max(const float* __restrict__ el,
                             const float* __restrict__ er,
                             const long long* __restrict__ src,
                             const long long* __restrict__ dst,
                             float* __restrict__ emax, long long E) {
    long long i = (long long)blockIdx.x * blockDim.x + threadIdx.x;
    if (i >= E * HEADS) return;
    long long e = i >> 3;
    int h = (int)(i & 7);
    long long s = src[e], d = dst[e];
    float v = lrelu(el[s * HEADS + h] + er[d * HEADS + h]);
    atomicMaxFloat(&emax[d * HEADS + h], v);
}

__global__ void gat_edge_exp(const float* __restrict__ el,
                             const float* __restrict__ er,
                             const long long* __restrict__ src,
                             const long long* __restrict__ dst,
                             const float* __restrict__ emax,
                             float* __restrict__ alpha,
                             float* __restrict__ asum, long long E) {
    long long i = (long long)blockIdx.x * blockDim.x + threadIdx.x;
    if (i >= E * HEADS) return;
    long long e = i >> 3;
    int h = (int)(i & 7);
    long long s = src[e], d = dst[e];
    float v = lrelu(el[s * HEADS + h] + er[d * HEADS + h]);
    float a = __expf(v - emax[d * HEADS + h]);
    alpha[i] = a;
    atomicAdd(&asum[d * HEADS + h], a);
}

__global__ void gat_edge_norm(float* __restrict__ alpha,
                              const float* __restrict__ asum,
                              const long long* __restrict__ dst, long long E) {
    long long i = (long long)blockIdx.x * blockDim.x + threadIdx.x;
    if (i >= E * HEADS) return;
    long long e = i >> 3;
    int h = (int)(i & 7);
    alpha[i] = alpha[i] / asum[dst[e] * HEADS + h];
}

// ---- weighted scatter: acc[dst][c] += feat[src][c] * alpha[e][c/D] ---------
// float4 over channels: D (64 or 40) divisible by 4, so head is quad-uniform.

__global__ void gat_scatter4(const float4* __restrict__ feat4,
                             const float* __restrict__ alpha,
                             const long long* __restrict__ src,
                             const long long* __restrict__ dst,
                             float* __restrict__ acc,
                             long long E, int C4, int D) {
    long long i = (long long)blockIdx.x * blockDim.x + threadIdx.x;
    if (i >= E * (long long)C4) return;
    long long e = i / C4;
    int c4 = (int)(i % C4);
    int c = c4 << 2;
    int h = c / D;
    long long s = src[e], d = dst[e];
    float w = alpha[e * HEADS + h];
    float4 f = feat4[s * (long long)C4 + c4];
    float* dp = &acc[d * (long long)(C4 << 2) + c];
    atomicAdd(dp + 0, f.x * w);
    atomicAdd(dp + 1, f.y * w);
    atomicAdd(dp + 2, f.z * w);
    atomicAdd(dp + 3, f.w * w);
}

// ---- final: out[n][c] = mean_h(acc2[n][h][c] + b2[h][c]) -------------------

__global__ void gat_final_mean(const float* __restrict__ acc,
                               const float* __restrict__ b2,
                               float* __restrict__ out, int N, int D) {
    long long i = (long long)blockIdx.x * blockDim.x + threadIdx.x;
    if (i >= (long long)N * D) return;
    int n = (int)(i / D);
    int c = (int)(i % D);
    float s = 0.0f;
#pragma unroll
    for (int h = 0; h < HEADS; ++h)
        s += acc[(size_t)n * HEADS * D + h * D + c] + b2[h * D + c];
    out[i] = s * (1.0f / HEADS);
}

// ---------------------------------------------------------------------------

static inline size_t alignUp256(size_t x) { return (x + 255) & ~(size_t)255; }
static inline int blocksFor(long long n, int b) { return (int)((n + b - 1) / b); }

extern "C" void kernel_launch(void* const* d_in, const int* in_sizes, int n_in,
                              void* d_out, int out_size, void* d_ws, size_t ws_size,
                              hipStream_t stream) {
    (void)n_in; (void)out_size; (void)ws_size;

    const float*     features = (const float*)d_in[0];  // [N,512]
    const float*     W1  = (const float*)d_in[1];       // [512,512]
    const float*     al1 = (const float*)d_in[2];       // [8,64]
    const float*     ar1 = (const float*)d_in[3];
    const float*     b1  = (const float*)d_in[4];       // [512]
    const float*     W2  = (const float*)d_in[5];       // [512,320]
    const float*     al2 = (const float*)d_in[6];       // [8,40]
    const float*     ar2 = (const float*)d_in[7];
    const float*     b2  = (const float*)d_in[8];       // [320]
    const long long* src = (const long long*)d_in[9];   // int64 per reference
    const long long* dst = (const long long*)d_in[10];
    float* out = (float*)d_out;                         // [N,40]

    const int IN_FEATS = 512, HID = 64, NC = 40;
    const int C1 = HEADS * HID;   // 512
    const int C2 = HEADS * NC;    // 320
    const int N = in_sizes[0] / IN_FEATS;               // 100000
    const long long E = in_sizes[9];                    // 500000

    // ---- workspace carve ----
    char* p = (char*)d_ws;
    unsigned short* Xb   = (unsigned short*)p; p += alignUp256((size_t)N * IN_FEATS * 2); // reused as Hb
    unsigned short* W1t  = (unsigned short*)p; p += alignUp256((size_t)IN_FEATS * C1 * 2);
    unsigned short* W2t  = (unsigned short*)p; p += alignUp256((size_t)C1 * C2 * 2);
    float* feat = (float*)p; p += alignUp256((size_t)N * C1 * 4);   // feat1, reused feat2
    float* acc  = (float*)p; p += alignUp256((size_t)N * C1 * 4);   // acc1, reused acc2
    float* el   = (float*)p; p += alignUp256((size_t)N * HEADS * 4);
    float* er   = (float*)p; p += alignUp256((size_t)N * HEADS * 4);
    float* emax = (float*)p; p += alignUp256((size_t)N * HEADS * 4);
    float* asum = (float*)p; p += alignUp256((size_t)N * HEADS * 4);
    float* alpha= (float*)p; p += alignUp256((size_t)E * HEADS * 4);

    const int TB = 256;
    const long long NH = (long long)N * HEADS;
    const long long EH = E * HEADS;

    // ==================== Layer 1 ====================
    gat_cast_bf16<<<blocksFor((long long)N * IN_FEATS, TB), TB, 0, stream>>>(
        features, Xb, (long long)N * IN_FEATS);
    gat_transpose_cast<<<blocksFor((long long)IN_FEATS * C1, TB), TB, 0, stream>>>(
        W1, W1t, IN_FEATS, C1);

    {   // feat1 = X @ W1
        long long waves = (long long)(N / (16 * MT)) * (C1 / (16 * NT));
        gat_wmma_gemm<<<blocksFor(waves, 8), TB, 0, stream>>>(
            Xb, W1t, feat, N, IN_FEATS, C1);
    }

    gat_node_scores<<<blocksFor(N, 8), TB, 0, stream>>>(feat, al1, ar1, el, er, N, HID);

    gat_fill_f32<<<blocksFor(NH, TB), TB, 0, stream>>>(emax, -3.0e38f, NH);
    gat_fill_f32<<<blocksFor(NH, TB), TB, 0, stream>>>(asum, 0.0f, NH);
    gat_edge_max <<<blocksFor(EH, TB), TB, 0, stream>>>(el, er, src, dst, emax, E);
    gat_edge_exp <<<blocksFor(EH, TB), TB, 0, stream>>>(el, er, src, dst, emax, alpha, asum, E);
    gat_edge_norm<<<blocksFor(EH, TB), TB, 0, stream>>>(alpha, asum, dst, E);

    gat_fill_f32<<<blocksFor((long long)N * C1, TB), TB, 0, stream>>>(acc, 0.0f, (long long)N * C1);
    gat_scatter4<<<blocksFor(E * (C1 / 4), TB), TB, 0, stream>>>(
        (const float4*)feat, alpha, src, dst, acc, E, C1 / 4, HID);

    // h = relu(acc1 + b1) -> bf16 (reuse Xb)
    gat_relu_bias_cast<<<blocksFor((long long)N * C1, TB), TB, 0, stream>>>(
        acc, b1, C1, Xb, (long long)N * C1);

    // ==================== Layer 2 ====================
    gat_transpose_cast<<<blocksFor((long long)C1 * C2, TB), TB, 0, stream>>>(
        W2, W2t, C1, C2);

    {   // feat2 = H @ W2  (reuse feat buffer)
        long long waves = (long long)(N / (16 * MT)) * (C2 / (16 * NT));
        gat_wmma_gemm<<<blocksFor(waves, 8), TB, 0, stream>>>(
            Xb, W2t, feat, N, C1, C2);
    }

    gat_node_scores<<<blocksFor(N, 8), TB, 0, stream>>>(feat, al2, ar2, el, er, N, NC);

    gat_fill_f32<<<blocksFor(NH, TB), TB, 0, stream>>>(emax, -3.0e38f, NH);
    gat_fill_f32<<<blocksFor(NH, TB), TB, 0, stream>>>(asum, 0.0f, NH);
    gat_edge_max <<<blocksFor(EH, TB), TB, 0, stream>>>(el, er, src, dst, emax, E);
    gat_edge_exp <<<blocksFor(EH, TB), TB, 0, stream>>>(el, er, src, dst, emax, alpha, asum, E);
    gat_edge_norm<<<blocksFor(EH, TB), TB, 0, stream>>>(alpha, asum, dst, E);

    gat_fill_f32<<<blocksFor((long long)N * C2, TB), TB, 0, stream>>>(acc, 0.0f, (long long)N * C2);
    gat_scatter4<<<blocksFor(E * (C2 / 4), TB), TB, 0, stream>>>(
        (const float4*)feat, alpha, src, dst, acc, E, C2 / 4, NC);

    gat_final_mean<<<blocksFor((long long)N * NC, TB), TB, 0, stream>>>(acc, b2, out, N, NC);
}
